// AMMConv2d_7421703488261
// MI455X (gfx1250) — matrix operand order repaired and checked by reading, not compiled, not run on hardware
//
#include <hip/hip_runtime.h>
#include <hip/hip_bf16.h>

// ---------------------------------------------------------------------------
// AMMConv2d forward on gfx1250 (MI455X).
// Forward value of the STE expression is exactly `quantized + bias`, so:
//   1) lut[c,k,o]   = sum_s centroids[c,k,s] * weight[c,s,o]        (prep)
//   2) k*[n,c]      = argmin_k ||v[n,c]-cent[c,k]||  (== argmax v.c - 0.5|c|^2)
//                     -> 16x16 tiles via v_wmma_f32_16x16x32_bf16, K=72 pad 96
//   3) out[n,:]     = bias + sum_c lut[c, k*[n,c], :]               (gather)
// inverse_temperature_logit only affects gradients -> unused in forward.
// ---------------------------------------------------------------------------

#define NC   16
#define KK   16
#define SUB  72
#define CIN  128
#define COUT 256
#define HH   56
#define WW   56
#define OH   56
#define OW   56
#define BB   8
#define VROW 1544   // halves per row in vbuf: 16 cb * 96 + 8 pad (bank-skewed)

typedef __attribute__((ext_vector_type(16))) __bf16 v16bf;
typedef __attribute__((ext_vector_type(8)))  float  v8f;

union ABfrag {
    v16bf v;
    uint4 q[2];
};

__device__ __forceinline__ unsigned short f2bf(float f) {
    return (unsigned short)(__float_as_uint(f) >> 16);   // truncate to bf16
}

// ---------------------------------------------------------------------------
// Prep 1: lut[c*16+k][o] and c2half[c*16+k] = 0.5*|centroid|^2
// ---------------------------------------------------------------------------
__global__ __launch_bounds__(256)
void lut_kernel(const float* __restrict__ cent, const float* __restrict__ wgt,
                float* __restrict__ lut, float* __restrict__ c2half) {
    const int ck = blockIdx.x;              // 0..255 = cb*16 + k
    const int cb = ck >> 4;
    const int o  = threadIdx.x;             // 0..255
    const float* cr = cent + ck * SUB;
    const float* wb = wgt  + cb * SUB * COUT;
    float acc = 0.f;
    for (int s = 0; s < SUB; ++s)
        acc = fmaf(cr[s], wb[s * COUT + o], acc);
    lut[ck * COUT + o] = acc;
    if (o == 0) {
        float s2 = 0.f;
        for (int s = 0; s < SUB; ++s) s2 += cr[s] * cr[s];
        c2half[ck] = 0.5f * s2;
    }
}

// ---------------------------------------------------------------------------
// Prep 2: pre-swizzled bf16 B-fragments of the centroids, zero-padded K=96.
// B layout used here (contiguous k, same as A): lane L holds column n = L%16; half e holds
// K = 32*t + 16*(L/16) + e.   frag[(cb*3+t)*32 + L][e]
// ---------------------------------------------------------------------------
__global__ __launch_bounds__(32)
void bfrag_kernel(const float* __restrict__ cent, unsigned short* __restrict__ frag) {
    const int blk = blockIdx.x;             // 0..47 = cb*3 + t
    const int cb  = blk / 3, t = blk % 3;
    const int L   = threadIdx.x;            // 0..31
    unsigned short* dst = frag + (blk * 32 + L) * 16;
    const int n     = L & 15;
    const int kbase = 32 * t + ((L >> 4) << 4);
    for (int e = 0; e < 16; ++e) {
        int s = kbase + e;
        float v = (s < SUB) ? cent[(cb * 16 + n) * SUB + s] : 0.f;
        dst[e] = f2bf(v);
    }
}

// ---------------------------------------------------------------------------
// Stage B: per-position, per-codebook argmin via WMMA.
// Block = (b, oh, 16-wide ow tile); 4 waves x 4 codebooks each.
// vbuf[row][cb*96 + s] holds bf16 of the unfolded subvector, zero-padded to
// K=96, so each WMMA A-fragment is exactly two contiguous ds_load_b128.
// ---------------------------------------------------------------------------
__global__ __launch_bounds__(128)
void argmin_kernel(const float* __restrict__ x, const unsigned short* __restrict__ frag,
                   const float* __restrict__ c2half, unsigned char* __restrict__ idxbuf) {
    __shared__ float          xs[CIN * 54];      // [ch][kh(3)][col(18)] 27.6 KB
    __shared__ unsigned short vbuf[16 * VROW];   // [row][cb*96+s]       49.4 KB

    const int bx   = blockIdx.x;
    const int tile = bx & 3;
    const int ohh  = (bx >> 2) % OH;
    const int b    = bx / (4 * OH);
    const int ow0  = tile * 16;
    const int valid = (OW - ow0 < 16) ? (OW - ow0) : 16;

    // ---- stage 1: global -> LDS x tile, zero-padded halo (coalesced) ----
    for (int i = threadIdx.x; i < CIN * 54; i += 128) {
        int ch  = i / 54, rem = i % 54;
        int r   = rem / 18, col = rem % 18;
        int h   = ohh + r - 1, w = ow0 + col - 1;
        float v = 0.f;
        if ((unsigned)h < (unsigned)HH && (unsigned)w < (unsigned)WW)
            v = x[((b * CIN + ch) * HH + h) * WW + w];
        xs[i] = v;
    }
    __syncthreads();

    // ---- stage 2a: zero the K=72..95 pad of every (row, cb) ----
    for (int i = threadIdx.x; i < 16 * NC * 12; i += 128) {       // 12 dwords pad
        int row = i / (NC * 12), rem = i % (NC * 12);
        int cb  = rem / 12,      q   = rem % 12;
        ((unsigned int*)(vbuf + row * VROW + cb * 96 + 72))[q] = 0u;
    }
    // ---- stage 2b: branch-free gather + bf16 convert into A-layout ----
    for (int i = threadIdx.x; i < 16 * CIN; i += 128) {
        int row = i >> 7, ch = i & (CIN - 1);
        const float* px = xs + ch * 54 + row;
        unsigned short* dst = vbuf + row * VROW + (ch >> 3) * 96 + (ch & 7) * 9;
        #pragma unroll
        for (int kh = 0; kh < 3; ++kh) {
            float w0 = px[kh * 18 + 0];
            float w1 = px[kh * 18 + 1];
            float w2 = px[kh * 18 + 2];
            dst[kh * 3 + 0] = f2bf(w0);
            dst[kh * 3 + 1] = f2bf(w1);
            dst[kh * 3 + 2] = f2bf(w2);
        }
    }
    __syncthreads();

    // ---- stage 3: WMMA argmin ----
    const int L     = threadIdx.x & 31;
    const int wv    = threadIdx.x >> 5;
    const int row16 = L & 15;               // A: row M / C: column N (centroid k)
    const int hi    = L >> 4;
    const int nbase = (b * OH + ohh) * OW + ow0;

    for (int cc = 0; cc < 4; ++cc) {
        const int cb    = wv + cc * 4;
        const float c2v = c2half[cb * 16 + row16];

        const unsigned short* arow = vbuf + row16 * VROW + cb * 96 + 16 * hi;

        v8f acc = {};
        #pragma unroll
        for (int t = 0; t < 3; ++t) {
            ABfrag a, bm;
            a.q[0] = *(const uint4*)(arow + 32 * t);
            a.q[1] = *(const uint4*)(arow + 32 * t + 8);
            const uint4* bpt = (const uint4*)(frag + ((cb * 3 + t) * 32 + L) * 16);
            bm.q[0] = bpt[0];
            bm.q[1] = bpt[1];
            acc = __builtin_amdgcn_wmma_f32_16x16x32_bf16(
                      false, a.v, false, bm.v, (short)0, acc, false, false);
        }

        // score = v.c_k - 0.5|c_k|^2 ; branchless argmax over the 16 k-lanes.
        // Order-preserving uint key, low 4 bits = (15-k) so ties pick low k.
        #pragma unroll
        for (int r = 0; r < 8; ++r) {
            float sc   = acc[r] - c2v;
            unsigned u = __float_as_uint(sc);
            u ^= (unsigned)(((int)u) >> 31) | 0x80000000u;
            unsigned key = (u & 0xFFFFFFF0u) | (15u - (unsigned)row16);
            #pragma unroll
            for (int off = 8; off >= 1; off >>= 1) {
                unsigned ok = (unsigned)__shfl_xor((int)key, off, 16);
                key = (key > ok) ? key : ok;          // v_max_u32
            }
            int rw = r + 8 * hi;                      // C layout: M = r (+8 hi half)
            if (row16 == 0 && rw < valid)
                idxbuf[(nbase + rw) * NC + cb] = (unsigned char)(15u - (key & 15u));
        }
    }
}

// ---------------------------------------------------------------------------
// Stage C: out[b,o,oh,ow] = bias[o] + sum_c lut[c, k*, o]; LDS transpose so
// every thread streams contiguous 128B float4 segments of the NCHW output.
// Block = 32 positions x 256 channels; 8 waves x 4 rows each.
// ---------------------------------------------------------------------------
__global__ __launch_bounds__(256)
void gather_kernel(const unsigned char* __restrict__ idxbuf, const float* __restrict__ lut,
                   const float* __restrict__ bias, float* __restrict__ out) {
    __shared__ float        outs[COUT * 32];          // [o][row] 32 KB
    __shared__ unsigned int idxs[128];                // 32 rows x 16 bytes

    const int n0  = blockIdx.x * 32;
    const int tid = threadIdx.x;
    if (tid < 128)
        idxs[tid] = ((const unsigned int*)idxbuf)[n0 * 4 + tid];
    __syncthreads();

    const int L  = tid & 31;
    const int wv = tid >> 5;
    const unsigned char* idc  = (const unsigned char*)idxs;
    const float4* lut4  = (const float4*)lut;
    const float4* bias4 = (const float4*)bias;

    for (int rr = 0; rr < 4; ++rr) {
        const int row = wv * 4 + rr;
        float4 a0 = bias4[L];                 // o = 4L .. 4L+3
        float4 a1 = bias4[L + 32];            // o = 128+4L ..
        const unsigned char* id = idc + row * NC;
        #pragma unroll
        for (int c = 0; c < NC; ++c) {
            const float4* lp = lut4 + (c * KK + (int)id[c]) * (COUT / 4);
            float4 v0 = lp[L], v1 = lp[L + 32];
            a0.x += v0.x; a0.y += v0.y; a0.z += v0.z; a0.w += v0.w;
            a1.x += v1.x; a1.y += v1.y; a1.z += v1.z; a1.w += v1.w;
        }
        outs[(4 * L + 0) * 32 + row] = a0.x;
        outs[(4 * L + 1) * 32 + row] = a0.y;
        outs[(4 * L + 2) * 32 + row] = a0.z;
        outs[(4 * L + 3) * 32 + row] = a0.w;
        outs[(128 + 4 * L + 0) * 32 + row] = a1.x;
        outs[(128 + 4 * L + 1) * 32 + row] = a1.y;
        outs[(128 + 4 * L + 2) * 32 + row] = a1.z;
        outs[(128 + 4 * L + 3) * 32 + row] = a1.w;
    }
    __syncthreads();

    // writeout: thread = output channel o; 32 positions never straddle a batch
    const int o   = tid;
    const int b   = n0 / (OH * OW);
    const int rem = n0 % (OH * OW);
    float4* dst = (float4*)(out + ((size_t)b * COUT + o) * (OH * OW) + rem);
    const float4* src = (const float4*)(outs + o * 32);
    #pragma unroll
    for (int q = 0; q < 8; ++q) dst[q] = src[q];
}

// ---------------------------------------------------------------------------
extern "C" void kernel_launch(void* const* d_in, const int* in_sizes, int n_in,
                              void* d_out, int out_size, void* d_ws, size_t ws_size,
                              hipStream_t stream) {
    (void)in_sizes; (void)n_in; (void)out_size; (void)ws_size;
    const float* x    = (const float*)d_in[0];   // [8,128,56,56]
    const float* cent = (const float*)d_in[1];   // [16,16,72]
    const float* wgt  = (const float*)d_in[2];   // [16,72,256]
    // d_in[3] = inverse_temperature_logit: gradient-only, unused in forward
    const float* bias = (const float*)d_in[4];   // [256]
    float* out = (float*)d_out;

    char* ws = (char*)d_ws;
    float*          lut    = (float*)(ws);                       // 256*256*4 = 262144
    float*          c2half = (float*)(ws + 262144);              // 256*4     = 1024
    unsigned short* frag   = (unsigned short*)(ws + 263168);     // 48*32*16*2= 49152
    unsigned char*  idxbuf = (unsigned char*)(ws + 312320);      // 25088*16  = 401408

    lut_kernel   <<<NC * KK,        256, 0, stream>>>(cent, wgt, lut, c2half);
    bfrag_kernel <<<NC * 3,          32, 0, stream>>>(cent, frag);
    argmin_kernel<<<BB * OH * 4,    128, 0, stream>>>(x, frag, c2half, idxbuf);
    gather_kernel<<<BB * OH * OW / 32, 256, 0, stream>>>(idxbuf, lut, bias, out);
}
